// Sparse_Transformer_77867757076820
// MI455X (gfx1250) — matrix operand order repaired and checked by reading, not compiled, not run on hardware
//
#include <hip/hip_runtime.h>
#include <hip/hip_bf16.h>

// ---- problem dims (fixed by reference) ----
#define NN_HID 128
#define NN_C   256           // h width
#define NN_H   4             // heads
#define NN_OC  64            // per-head out channels
#define NN_FF  512           // ffn hidden
#define LN_EPS 1e-5f
#define SLOPE  0.2f

typedef __attribute__((ext_vector_type(16))) __bf16 v16bf;
typedef __attribute__((ext_vector_type(8)))  __bf16 v8bf;
typedef __attribute__((ext_vector_type(8)))  float  v8f;

// ---------------------------------------------------------------------------
// WMMA GEMM:  C[M,Nout] = A[M,K](bf16) * Bt[Nout,K](bf16, pre-transposed)
// Each wave computes a 32x64 tile: 2 M-subtiles x 4 N-subtiles, K-step 32.
// A fragment (16-bit A 16x32, contiguous k order used by this code): lanes r and r+16 both hold row r;
//   half=lane>>4 selects K groups {0..15} vs {16..31}
//   -> two contiguous 16B loads per lane per K-step.
// B fragment: lane holds column n, K=k0+half*16 .. +15, contiguous in the
//   [Nout,K] transposed weight -> one 32B load per lane per K-step.
// ---------------------------------------------------------------------------
template<bool RELU>
__global__ __launch_bounds__(256)
void gemm_bf16_wmma(const __bf16* __restrict__ A, const __bf16* __restrict__ Bt,
                    const float* __restrict__ bias, float* __restrict__ C,
                    __bf16* __restrict__ Cb,   // optional bf16 mirror of output
                    int M, int K, int Nout)
{
    const int lane = threadIdx.x & 31;
    const int wave = blockIdx.x * (blockDim.x >> 5) + (threadIdx.x >> 5);
    const int tilesN = Nout >> 6;   // 64-wide N tiles
    const int tilesM = M >> 5;      // 32-high M tiles
    if (wave >= tilesM * tilesN) return;
    const int tm = wave / tilesN;
    const int tn = wave % tilesN;
    const int r    = lane & 15;
    const int half = lane >> 4;

    const __bf16* arow0 = A + (size_t)(tm * 32 + r) * K + half * 16;
    const __bf16* arow1 = arow0 + (size_t)16 * K;
    const int colbase = tn << 6;

    const v8f vz = {0.f,0.f,0.f,0.f,0.f,0.f,0.f,0.f};
    v8f acc[2][4];
#pragma unroll
    for (int m = 0; m < 2; ++m)
#pragma unroll
        for (int t = 0; t < 4; ++t) acc[m][t] = vz;

    const __bf16* bcol[4];
#pragma unroll
    for (int t = 0; t < 4; ++t)
        bcol[t] = Bt + (size_t)(colbase + t * 16 + r) * K + half * 16;

    for (int k0 = 0; k0 < K; k0 += 32) {
        const __bf16* p0 = arow0 + k0;
        const __bf16* p1 = arow1 + k0;
        v8bf lo0 = *(const v8bf*)(p0);
        v8bf hi0 = *(const v8bf*)(p0 + 8);
        v8bf lo1 = *(const v8bf*)(p1);
        v8bf hi1 = *(const v8bf*)(p1 + 8);
        v16bf a0 = __builtin_shufflevector(lo0, hi0, 0,1,2,3,4,5,6,7,8,9,10,11,12,13,14,15);
        v16bf a1 = __builtin_shufflevector(lo1, hi1, 0,1,2,3,4,5,6,7,8,9,10,11,12,13,14,15);

        v16bf b[4];
#pragma unroll
        for (int t = 0; t < 4; ++t)
            b[t] = *(const v16bf*)(bcol[t] + k0);

        if (k0 + 32 < K) {                 // global_prefetch_b8 for next K-slab
            __builtin_prefetch(p0 + 32, 0, 1);
            __builtin_prefetch(p1 + 32, 0, 1);
        }
#pragma unroll
        for (int t = 0; t < 4; ++t) {
            acc[0][t] = __builtin_amdgcn_wmma_f32_16x16x32_bf16(
                false, a0, false, b[t], (short)0, acc[0][t], false, false);
            acc[1][t] = __builtin_amdgcn_wmma_f32_16x16x32_bf16(
                false, a1, false, b[t], (short)0, acc[1][t], false, false);
        }
    }

    // C/D layout: vgpr j, lane<16 -> (M=j, N=lane); lane>=16 -> (M=8+j, N=lane-16)
#pragma unroll
    for (int m = 0; m < 2; ++m) {
#pragma unroll
        for (int t = 0; t < 4; ++t) {
            const int col = colbase + t * 16 + r;
            const float bv = bias ? bias[col] : 0.f;
#pragma unroll
            for (int j = 0; j < 8; ++j) {
                const int row = tm * 32 + m * 16 + half * 8 + j;
                float v = acc[m][t][j] + bv;
                if (RELU) v = v > 0.f ? v : 0.f;
                C[(size_t)row * Nout + col] = v;
                if (Cb) Cb[(size_t)row * Nout + col] = (__bf16)v;
            }
        }
    }
}

// ---- helpers -------------------------------------------------------------

__global__ void concat_kernel(const float* __restrict__ hf, const float* __restrict__ hs,
                              float* __restrict__ h, __bf16* __restrict__ hb, int N)
{
    int i = blockIdx.x * blockDim.x + threadIdx.x;
    if (i >= N * NN_C) return;
    int n = i / NN_C, c = i % NN_C;
    float v = (c < NN_HID) ? hf[n * NN_HID + c] : hs[n * NN_HID + (c - NN_HID)];
    h[i]  = v;
    hb[i] = (__bf16)v;
}

// src: [L, K, Nout] f32  ->  dst: [L, Nout, K] bf16
__global__ void transpose_to_bf16(const float* __restrict__ src, __bf16* __restrict__ dst,
                                  int K, int Nout, int L)
{
    int i = blockIdx.x * blockDim.x + threadIdx.x;
    int per = K * Nout;
    if (i >= L * per) return;
    int l = i / per, rem = i % per;
    int n = rem / K, k = rem % K;
    dst[i] = (__bf16)src[(size_t)l * per + (size_t)k * Nout + n];
}

__global__ void attn_scores(const float* __restrict__ xh, const float* __restrict__ asrc,
                            const float* __restrict__ adst, float* __restrict__ a_s,
                            float* __restrict__ a_d, int N)
{
    int i = blockIdx.x * blockDim.x + threadIdx.x;      // n*H + h
    if (i >= N * NN_H) return;
    int n = i / NN_H, h = i % NN_H;
    const float* x  = xh   + (size_t)n * NN_C + h * NN_OC;
    const float* ws = asrc + h * NN_OC;
    const float* wd = adst + h * NN_OC;
    float ss = 0.f, sd = 0.f;
#pragma unroll 8
    for (int c = 0; c < NN_OC; ++c) { float v = x[c]; ss += v * ws[c]; sd += v * wd[c]; }
    a_s[i] = ss; a_d[i] = sd;
}

__global__ void fill_kernel(float* __restrict__ p, float v, int n)
{
    int i = blockIdx.x * blockDim.x + threadIdx.x;
    if (i < n) p[i] = v;
}

__device__ inline void atomicMaxF(float* addr, float val)
{
    int* ia = (int*)addr;
    int cur = *ia;
    while (__int_as_float(cur) < val) {
        int prev = atomicCAS(ia, cur, __float_as_int(val));
        if (prev == cur) break;
        cur = prev;
    }
}

__global__ void edge_max_kernel(const int* __restrict__ ei, const float* __restrict__ a_s,
                                const float* __restrict__ a_d, float* __restrict__ ebuf,
                                float* __restrict__ m, int E, int N)
{
    int e = blockIdx.x * blockDim.x + threadIdx.x;
    if (e >= E + N) return;
    int src = (e < E) ? ei[e]     : (e - E);
    int dst = (e < E) ? ei[E + e] : (e - E);
#pragma unroll
    for (int h = 0; h < NN_H; ++h) {
        float s = a_s[src * NN_H + h] + a_d[dst * NN_H + h];
        s = (s > 0.f) ? s : SLOPE * s;                 // leaky_relu
        ebuf[(size_t)e * NN_H + h] = s;
        atomicMaxF(&m[dst * NN_H + h], s);
    }
}

__global__ void edge_soft_kernel(const int* __restrict__ ei, float* __restrict__ ebuf,
                                 const float* __restrict__ m, float* __restrict__ z,
                                 int E, int N)
{
    int e = blockIdx.x * blockDim.x + threadIdx.x;
    if (e >= E + N) return;
    int dst = (e < E) ? ei[E + e] : (e - E);
#pragma unroll
    for (int h = 0; h < NN_H; ++h) {
        float ex = __expf(ebuf[(size_t)e * NN_H + h] - m[dst * NN_H + h]);
        ebuf[(size_t)e * NN_H + h] = ex;
        atomicAdd(&z[dst * NN_H + h], ex);
    }
}

__global__ void edge_msg_kernel(const int* __restrict__ ei, const float* __restrict__ ebuf,
                                const float* __restrict__ z, const float* __restrict__ xh,
                                float* __restrict__ agg, int E, int N)
{
    int i = blockIdx.x * blockDim.x + threadIdx.x;      // (E+N)*H
    if (i >= (E + N) * NN_H) return;
    int e = i >> 2, h = i & 3;
    int src = (e < E) ? ei[e]     : (e - E);
    int dst = (e < E) ? ei[E + e] : (e - E);
    float alpha = ebuf[i] / z[dst * NN_H + h];
    const float4* x = (const float4*)(xh + (size_t)src * NN_C + h * NN_OC);
    float* o = agg + (size_t)dst * NN_C + h * NN_OC;
#pragma unroll
    for (int c = 0; c < NN_OC / 4; ++c) {
        float4 v = x[c];
        atomicAdd(o + 4 * c + 0, v.x * alpha);
        atomicAdd(o + 4 * c + 1, v.y * alpha);
        atomicAdd(o + 4 * c + 2, v.z * alpha);
        atomicAdd(o + 4 * c + 3, v.w * alpha);
    }
}

// out[row] = LN(x[row] + resid[row] + bias) * g + b   (wave per row; C=256, 8/lane)
// also writes bf16 mirror for the next WMMA GEMM.
__global__ __launch_bounds__(256)
void ln_kernel(const float* __restrict__ x, const float* __restrict__ resid,
               const float* __restrict__ bias, const float* __restrict__ g,
               const float* __restrict__ b, float* __restrict__ out,
               __bf16* __restrict__ outb, int N)
{
    int lane = threadIdx.x & 31;
    int row  = blockIdx.x * (blockDim.x >> 5) + (threadIdx.x >> 5);
    if (row >= N) return;
    const float* xr = x     + (size_t)row * NN_C;
    const float* rr = resid + (size_t)row * NN_C;
    float v[8];
    float s = 0.f;
#pragma unroll
    for (int i = 0; i < 8; ++i) {
        int c = lane + 32 * i;
        float t = xr[c] + rr[c] + (bias ? bias[c] : 0.f);
        v[i] = t; s += t;
    }
#pragma unroll
    for (int o = 16; o > 0; o >>= 1) s += __shfl_xor(s, o, 32);
    float mean = s * (1.f / NN_C);
    float q = 0.f;
#pragma unroll
    for (int i = 0; i < 8; ++i) { float d = v[i] - mean; q += d * d; }
#pragma unroll
    for (int o = 16; o > 0; o >>= 1) q += __shfl_xor(q, o, 32);
    float inv = rsqrtf(q * (1.f / NN_C) + LN_EPS);
#pragma unroll
    for (int i = 0; i < 8; ++i) {
        int c = lane + 32 * i;
        float y = (v[i] - mean) * inv * g[c] + b[c];
        out[(size_t)row * NN_C + c]  = y;
        outb[(size_t)row * NN_C + c] = (__bf16)y;
    }
}

__global__ void split_kernel(const float* __restrict__ h, float* __restrict__ out, int N)
{
    int i = blockIdx.x * blockDim.x + threadIdx.x;
    if (i >= N * NN_C) return;
    int n = i / NN_C, c = i % NN_C;
    if (c < NN_HID) out[n * NN_HID + c] = h[i];
    else            out[(size_t)N * NN_HID + n * NN_HID + (c - NN_HID)] = h[i];
}

// ---------------------------------------------------------------------------
extern "C" void kernel_launch(void* const* d_in, const int* in_sizes, int n_in,
                              void* d_out, int out_size, void* d_ws, size_t ws_size,
                              hipStream_t stream)
{
    const float* hf      = (const float*)d_in[0];
    const float* hs      = (const float*)d_in[1];
    const int*   ei      = (const int*)  d_in[2];
    const float* W       = (const float*)d_in[3];
    const float* att_src = (const float*)d_in[4];
    const float* att_dst = (const float*)d_in[5];
    const float* gat_b   = (const float*)d_in[6];
    const float* w1      = (const float*)d_in[7];
    const float* b1      = (const float*)d_in[8];
    const float* w2      = (const float*)d_in[9];
    const float* b2      = (const float*)d_in[10];
    const float* ln1g    = (const float*)d_in[11];
    const float* ln1b    = (const float*)d_in[12];
    const float* ln2g    = (const float*)d_in[13];
    const float* ln2b    = (const float*)d_in[14];

    const int N = in_sizes[0] / NN_HID;        // 20000
    const int E = in_sizes[2] / 2;             // 320000
    const int L = in_sizes[3] / (NN_C * NN_C); // 12

    // ---- workspace carve (256B aligned) ----
    char* wsp = (char*)d_ws;
    auto carve = [&](size_t bytes) {
        char* p = wsp;
        wsp += (bytes + 255) & ~(size_t)255;
        return (void*)p;
    };
    float*  h    = (float*) carve((size_t)N * NN_C  * 4);
    float*  xh   = (float*) carve((size_t)N * NN_C  * 4);   // reused as f2
    float*  agg  = (float*) carve((size_t)N * NN_C  * 4);
    float*  f1   = (float*) carve((size_t)N * NN_FF * 4);
    __bf16* hb   = (__bf16*)carve((size_t)N * NN_C  * 2);   // bf16 mirror of h
    __bf16* f1b  = (__bf16*)carve((size_t)N * NN_FF * 2);   // bf16 mirror of f1
    float*  a_s  = (float*) carve((size_t)N * NN_H  * 4);
    float*  a_d  = (float*) carve((size_t)N * NN_H  * 4);
    float*  mbuf = (float*) carve((size_t)N * NN_H  * 4);
    float*  zbuf = (float*) carve((size_t)N * NN_H  * 4);
    float*  ebuf = (float*) carve((size_t)(E + N) * NN_H * 4);
    __bf16* Wt   = (__bf16*)carve((size_t)L * NN_C  * NN_C  * 2);
    __bf16* w1t  = (__bf16*)carve((size_t)L * NN_C  * NN_FF * 2);
    __bf16* w2t  = (__bf16*)carve((size_t)L * NN_FF * NN_C  * 2);

    const int TB = 256;
    auto blk = [](int n, int t) { return (n + t - 1) / t; };

    // ---- one-time per call: concat inputs, convert+transpose weights ----
    concat_kernel<<<blk(N * NN_C, TB), TB, 0, stream>>>(hf, hs, h, hb, N);
    transpose_to_bf16<<<blk(L * NN_C * NN_C,  TB), TB, 0, stream>>>(W,  Wt,  NN_C,  NN_C,  L);
    transpose_to_bf16<<<blk(L * NN_C * NN_FF, TB), TB, 0, stream>>>(w1, w1t, NN_C,  NN_FF, L);
    transpose_to_bf16<<<blk(L * NN_FF * NN_C, TB), TB, 0, stream>>>(w2, w2t, NN_FF, NN_C,  L);

    const int EN = E + N;
    for (int l = 0; l < L; ++l) {
        // --- GATConv: xh = h @ W[l] (bf16 WMMA) ---
        {
            int waves = (N / 32) * (NN_C / 64);
            gemm_bf16_wmma<false><<<blk(waves * 32, TB), TB, 0, stream>>>(
                hb, Wt + (size_t)l * NN_C * NN_C, nullptr, xh, nullptr, N, NN_C, NN_C);
        }
        attn_scores<<<blk(N * NN_H, TB), TB, 0, stream>>>(
            xh, att_src + (size_t)l * NN_H * NN_OC, att_dst + (size_t)l * NN_H * NN_OC,
            a_s, a_d, N);

        fill_kernel<<<blk(N * NN_H, TB), TB, 0, stream>>>(mbuf, -1e30f, N * NN_H);
        fill_kernel<<<blk(N * NN_H, TB), TB, 0, stream>>>(zbuf, 0.f,    N * NN_H);
        fill_kernel<<<blk(N * NN_C, TB), TB, 0, stream>>>(agg,  0.f,    N * NN_C);

        edge_max_kernel <<<blk(EN,        TB), TB, 0, stream>>>(ei, a_s, a_d, ebuf, mbuf, E, N);
        edge_soft_kernel<<<blk(EN,        TB), TB, 0, stream>>>(ei, ebuf, mbuf, zbuf, E, N);
        edge_msg_kernel <<<blk(EN * NN_H, TB), TB, 0, stream>>>(ei, ebuf, zbuf, xh, agg, E, N);

        // h = LN(agg + gat_b + h)   (also refresh bf16 mirror hb)
        ln_kernel<<<blk(N * 32, TB), TB, 0, stream>>>(
            agg, h, gat_b + (size_t)l * NN_C,
            ln1g + (size_t)l * NN_C, ln1b + (size_t)l * NN_C, h, hb, N);

        // --- FFN: f1 = relu(h@w1+b1); f2 = f1@w2+b2; h = LN(f2 + h) ---
        {
            int waves = (N / 32) * (NN_FF / 64);
            gemm_bf16_wmma<true><<<blk(waves * 32, TB), TB, 0, stream>>>(
                hb, w1t + (size_t)l * NN_C * NN_FF, b1 + (size_t)l * NN_FF,
                f1, f1b, N, NN_C, NN_FF);
        }
        {
            int waves = (N / 32) * (NN_C / 64);
            gemm_bf16_wmma<false><<<blk(waves * 32, TB), TB, 0, stream>>>(
                f1b, w2t + (size_t)l * NN_FF * NN_C, b2 + (size_t)l * NN_C,
                xh, nullptr, N, NN_FF, NN_C);   // xh reused as f2
        }
        ln_kernel<<<blk(N * 32, TB), TB, 0, stream>>>(
            xh, h, nullptr, ln2g + (size_t)l * NN_C, ln2b + (size_t)l * NN_C, h, hb, N);
    }

    split_kernel<<<blk(N * NN_C, TB), TB, 0, stream>>>(h, (float*)d_out, N);
}
